// SVDHead_55009941127305
// MI455X (gfx1250) — matrix operand order repaired and checked by reading, not compiled, not run on hardware
//
#include <hip/hip_runtime.h>

#define B_  16
#define N_  2048
#define DK_ 512

typedef __attribute__((ext_vector_type(16))) _Float16 v16h;
typedef __attribute__((ext_vector_type(8)))  _Float16 v8h;
typedef __attribute__((ext_vector_type(4)))  _Float16 v4h;
typedef __attribute__((ext_vector_type(8)))  float    v8f;
typedef __attribute__((ext_vector_type(4)))  float    v4f;

// ---------------------------------------------------------------------------
// Kernel 1: convert both embeddings f32 -> f16 (vectorized 4-wide)
// ---------------------------------------------------------------------------
__global__ void svd_convert_f16(const float* __restrict__ srcE,
                                const float* __restrict__ tgtE,
                                _Float16* __restrict__ Qh,
                                _Float16* __restrict__ Kh,
                                int n4) {
  int i = blockIdx.x * blockDim.x + threadIdx.x;
  if (i >= n4) return;
  v4f a = ((const v4f*)srcE)[i];
  v4f b = ((const v4f*)tgtE)[i];
  v4h ah, bh;
  ah[0] = (_Float16)a[0]; ah[1] = (_Float16)a[1];
  ah[2] = (_Float16)a[2]; ah[3] = (_Float16)a[3];
  bh[0] = (_Float16)b[0]; bh[1] = (_Float16)b[1];
  bh[2] = (_Float16)b[2]; bh[3] = (_Float16)b[3];
  ((v4h*)Qh)[i] = ah;
  ((v4h*)Kh)[i] = bh;
}

// ---------------------------------------------------------------------------
// Kernel 2: fused flash attention with d_v = 3 (src points are the values).
// One wave32 per 16-row Q tile; Q resident in registers (16 A-fragments).
// Lane-LOCAL online softmax: no cross-lane ops inside the streaming loop;
// lane maxima are merged once at the end (exact, numerically stable).
// ---------------------------------------------------------------------------
__global__ __launch_bounds__(256, 1)
void svd_attn_corr(const _Float16* __restrict__ Qh,
                   const _Float16* __restrict__ Kh,
                   const float* __restrict__ srcPts,
                   float* __restrict__ corr) {
  const int wave  = threadIdx.x >> 5;
  const int lane  = threadIdx.x & 31;
  const int tile  = blockIdx.x * 8 + wave;   // 0 .. B*N/16-1 = 2047
  const int b     = tile >> 7;               // tile / (N/16)
  const int qBase = (tile & 127) << 4;       // 16-row tile base
  const int hi    = lane >> 4;               // lane half (0/1)
  const int lr    = lane & 15;               // row (A) / column (B) index

  // --- load resident Q tile: A-fragment layout (16x32 f16 per k-chunk) ---
  const _Float16* qrow = Qh + ((size_t)(b * N_ + qBase + lr)) * DK_;
  v16h q[16];
#pragma unroll
  for (int kc = 0; kc < 16; ++kc) {
    union { v16h v; v8h h[2]; } u;
    u.h[0] = *(const v8h*)(qrow + kc * 32 + hi * 8);
    u.h[1] = *(const v8h*)(qrow + kc * 32 + 16 + hi * 8);
    q[kc] = u.v;
  }

  // lane-local softmax state (log2 domain): rows i + 8*hi of this tile
  float m[8], l[8], acc[8][3];
#pragma unroll
  for (int i = 0; i < 8; ++i) {
    m[i] = -1e30f; l[i] = 0.0f;
    acc[i][0] = 0.0f; acc[i][1] = 0.0f; acc[i][2] = 0.0f;
  }

  // 1/sqrt(512) * log2(e): score -> log2-domain in one multiply
  const float SC2 = 0.04419417382415922f * 1.4426950408889634f;

  for (int kBase = 0; kBase < N_; kBase += 16) {
    // B-fragment source row (target row = kBase + lr), lane half picks k+8
    const _Float16* krow =
        Kh + ((size_t)(b * N_ + kBase + lr)) * DK_ + hi * 8;

    v8f c = {};
#pragma unroll
    for (int kc = 0; kc < 16; ++kc) {
      union { v16h v; v8h h[2]; } u;
      u.h[0] = *(const v8h*)(krow + kc * 32);
      u.h[1] = *(const v8h*)(krow + kc * 32 + 16);
      c = __builtin_amdgcn_wmma_f32_16x16x32_f16(
          false, q[kc], false, u.v, (short)0, c, false, false);
    }

    // per-lane value vector (src point of this lane's column)
    const float* sp = srcPts + ((size_t)(b * N_ + kBase + lr)) * 3;
    const float vx = sp[0], vy = sp[1], vz = sp[2];

#pragma unroll
    for (int i = 0; i < 8; ++i) {
      float s     = c[i] * SC2;                // log2-domain score
      float mn    = fmaxf(m[i], s);
      float alpha = exp2f(m[i] - mn);          // ==1 unless new lane max
      float p     = exp2f(s - mn);
      m[i] = mn;
      l[i] = l[i] * alpha + p;
      acc[i][0] = acc[i][0] * alpha + p * vx;
      acc[i][1] = acc[i][1] * alpha + p * vy;
      acc[i][2] = acc[i][2] * alpha + p * vz;
    }
  }

  // merge lane-local states: global row max, rescale, reduce, write
#pragma unroll
  for (int i = 0; i < 8; ++i) {
    float M = m[i];
#pragma unroll
    for (int off = 1; off < 16; off <<= 1)
      M = fmaxf(M, __shfl_xor(M, off, 32));
    float sc = exp2f(m[i] - M);
    float lv = l[i] * sc;
    float a0 = acc[i][0] * sc;
    float a1 = acc[i][1] * sc;
    float a2 = acc[i][2] * sc;
#pragma unroll
    for (int off = 1; off < 16; off <<= 1) {
      lv += __shfl_xor(lv, off, 32);
      a0 += __shfl_xor(a0, off, 32);
      a1 += __shfl_xor(a1, off, 32);
      a2 += __shfl_xor(a2, off, 32);
    }
    if (lr == 0) {
      float inv = 1.0f / lv;
      float* o = corr + ((size_t)(b * N_ + qBase + i + 8 * hi)) * 3;
      o[0] = a0 * inv; o[1] = a1 * inv; o[2] = a2 * inv;
    }
  }
}

// ---------------------------------------------------------------------------
// Kernel 3: per-batch reduction + 3x3 SVD (Jacobi on H^T H) -> R, t
// one block (256 threads) per batch
// ---------------------------------------------------------------------------
__global__ void svd_reduce_solve(const float* __restrict__ tgtPts,
                                 const float* __restrict__ corr,
                                 float* __restrict__ out) {
  const int b   = blockIdx.x;
  const int tid = threadIdx.x;

  float v[15];
#pragma unroll
  for (int k = 0; k < 15; ++k) v[k] = 0.0f;

  for (int r = tid; r < N_; r += 256) {
    const float* tp = tgtPts + ((size_t)(b * N_ + r)) * 3;
    const float* cp = corr   + ((size_t)(b * N_ + r)) * 3;
    float t0 = tp[0], t1 = tp[1], t2 = tp[2];
    float c0 = cp[0], c1 = cp[1], c2 = cp[2];
    v[0] += t0; v[1] += t1; v[2] += t2;
    v[3] += c0; v[4] += c1; v[5] += c2;
    v[6]  += t0 * c0; v[7]  += t0 * c1; v[8]  += t0 * c2;
    v[9]  += t1 * c0; v[10] += t1 * c1; v[11] += t1 * c2;
    v[12] += t2 * c0; v[13] += t2 * c1; v[14] += t2 * c2;
  }

  // wave reduce, then cross-wave via LDS
#pragma unroll
  for (int k = 0; k < 15; ++k)
#pragma unroll
    for (int off = 1; off < 32; off <<= 1)
      v[k] += __shfl_xor(v[k], off, 32);

  __shared__ float sh[8][15];
  if ((tid & 31) == 0)
#pragma unroll
    for (int k = 0; k < 15; ++k) sh[tid >> 5][k] = v[k];
  __syncthreads();

  if (tid != 0) return;

  float tot[15];
#pragma unroll
  for (int k = 0; k < 15; ++k) {
    float s = 0.0f;
    for (int w = 0; w < 8; ++w) s += sh[w][k];
    tot[k] = s;
  }

  const float invN = 1.0f / (float)N_;
  float mu_t[3] = { tot[0] * invN, tot[1] * invN, tot[2] * invN };
  float mu_c[3] = { tot[3] * invN, tot[4] * invN, tot[5] * invN };

  // H[i][j] = sum t_i c_j - N * mu_t_i * mu_c_j
  float H[3][3];
  for (int i = 0; i < 3; ++i)
    for (int j = 0; j < 3; ++j)
      H[i][j] = tot[6 + i * 3 + j] - (float)N_ * mu_t[i] * mu_c[j];

  // A = H^T H (symmetric)
  float A[3][3];
  for (int i = 0; i < 3; ++i)
    for (int j = 0; j < 3; ++j)
      A[i][j] = H[0][i] * H[0][j] + H[1][i] * H[1][j] + H[2][i] * H[2][j];

  float V[3][3] = {{1,0,0},{0,1,0},{0,0,1}};

  // cyclic Jacobi
  const int PQ[3][2] = {{0,1},{0,2},{1,2}};
  for (int sweep = 0; sweep < 12; ++sweep) {
    for (int r = 0; r < 3; ++r) {
      int p = PQ[r][0], q = PQ[r][1];
      float apq = A[p][q];
      if (fabsf(apq) > 1e-24f) {
        float tau = (A[q][q] - A[p][p]) / (2.0f * apq);
        float tt  = (tau >= 0.0f ? 1.0f : -1.0f) /
                    (fabsf(tau) + sqrtf(1.0f + tau * tau));
        float cc  = 1.0f / sqrtf(1.0f + tt * tt);
        float ss  = tt * cc;
        for (int k = 0; k < 3; ++k) {            // A = A * J
          float akp = A[k][p], akq = A[k][q];
          A[k][p] = cc * akp - ss * akq;
          A[k][q] = ss * akp + cc * akq;
        }
        for (int k = 0; k < 3; ++k) {            // A = J^T * A
          float apk = A[p][k], aqk = A[q][k];
          A[p][k] = cc * apk - ss * aqk;
          A[q][k] = ss * apk + cc * aqk;
        }
        for (int k = 0; k < 3; ++k) {            // V = V * J
          float vkp = V[k][p], vkq = V[k][q];
          V[k][p] = cc * vkp - ss * vkq;
          V[k][q] = ss * vkp + cc * vkq;
        }
      }
    }
  }

  // sort eigenpairs descending (columns of V)
  float e[3] = { A[0][0], A[1][1], A[2][2] };
  for (int pass = 0; pass < 3; ++pass) {
    int i0 = (pass == 1) ? 1 : 0;
    int i1 = i0 + 1;
    if (e[i0] < e[i1]) {
      float te = e[i0]; e[i0] = e[i1]; e[i1] = te;
      for (int k = 0; k < 3; ++k) {
        float tv = V[k][i0]; V[k][i0] = V[k][i1]; V[k][i1] = tv;
      }
    }
  }

  // U columns: u0 = norm(H v0); u1 = orth(H v1); u2 = u0 x u1
  float u0[3], u1[3], u2[3];
  for (int i = 0; i < 3; ++i)
    u0[i] = H[i][0] * V[0][0] + H[i][1] * V[1][0] + H[i][2] * V[2][0];
  float n0 = sqrtf(u0[0]*u0[0] + u0[1]*u0[1] + u0[2]*u0[2]);
  n0 = fmaxf(n0, 1e-20f);
  for (int i = 0; i < 3; ++i) u0[i] /= n0;

  for (int i = 0; i < 3; ++i)
    u1[i] = H[i][0] * V[0][1] + H[i][1] * V[1][1] + H[i][2] * V[2][1];
  float d01 = u0[0]*u1[0] + u0[1]*u1[1] + u0[2]*u1[2];
  for (int i = 0; i < 3; ++i) u1[i] -= d01 * u0[i];
  float n1 = sqrtf(u1[0]*u1[0] + u1[1]*u1[1] + u1[2]*u1[2]);
  n1 = fmaxf(n1, 1e-20f);
  for (int i = 0; i < 3; ++i) u1[i] /= n1;

  u2[0] = u0[1]*u1[2] - u0[2]*u1[1];
  u2[1] = u0[2]*u1[0] - u0[0]*u1[2];
  u2[2] = u0[0]*u1[1] - u0[1]*u1[0];

  // det(U)=+1 by construction, so det(U Vh) = det(V)
  float detV = V[0][0]*(V[1][1]*V[2][2] - V[1][2]*V[2][1])
             - V[0][1]*(V[1][0]*V[2][2] - V[1][2]*V[2][0])
             + V[0][2]*(V[1][0]*V[2][1] - V[1][1]*V[2][0]);

  // R = U diag(1,1,detV) V^T
  float R[3][3];
  for (int i = 0; i < 3; ++i)
    for (int j = 0; j < 3; ++j)
      R[i][j] = u0[i]*V[j][0] + u1[i]*V[j][1] + detV * u2[i]*V[j][2];

  float* Ro = out + (size_t)b * 9;
  for (int i = 0; i < 3; ++i)
    for (int j = 0; j < 3; ++j) Ro[i * 3 + j] = R[i][j];

  float* to = out + (size_t)B_ * 9 + (size_t)b * 3;
  for (int i = 0; i < 3; ++i)
    to[i] = mu_t[i] - (R[i][0]*mu_c[0] + R[i][1]*mu_c[1] + R[i][2]*mu_c[2]);
}

// ---------------------------------------------------------------------------
extern "C" void kernel_launch(void* const* d_in, const int* in_sizes, int n_in,
                              void* d_out, int out_size, void* d_ws, size_t ws_size,
                              hipStream_t stream) {
  const float* srcE = (const float*)d_in[0];  // [B,N,DK]
  const float* tgtE = (const float*)d_in[1];  // [B,N,DK]
  const float* srcP = (const float*)d_in[2];  // [B,N,3]
  const float* tgtP = (const float*)d_in[3];  // [B,N,3]

  const size_t embElems = (size_t)B_ * N_ * DK_;      // 16,777,216
  _Float16* Qh  = (_Float16*)d_ws;
  _Float16* Kh  = Qh + embElems;
  float*    cor = (float*)(Kh + embElems);            // [B,N,3]

  const int n4 = (int)(embElems / 4);
  svd_convert_f16<<<n4 / 256, 256, 0, stream>>>(srcE, tgtE, Qh, Kh, n4);

  // B*N/16 = 2048 Q-tiles, 8 waves (256 threads) per block -> 256 blocks
  svd_attn_corr<<<256, 256, 0, stream>>>(Qh, Kh, srcP, cor);

  svd_reduce_solve<<<B_, 256, 0, stream>>>(tgtP, cor, (float*)d_out);
}
